// KLinearQ8_50611894616843
// MI455X (gfx1250) — matrix operand order repaired and checked by reading, not compiled, not run on hardware
//
#include <hip/hip_runtime.h>

typedef __attribute__((ext_vector_type(16))) _Float16 v16h;
typedef __attribute__((ext_vector_type(8)))  _Float16 v8h;
typedef __attribute__((ext_vector_type(8)))  float    v8f;
typedef __attribute__((ext_vector_type(4)))  int      v4i;

#define IN_F   4096
#define OUT_F  14336
#define TOKENS 32

// Kernel 1: fold per-in-feature scales into x, convert to f16.
// xs[t,i] = (f16)(x[t,i] * scales[i]); 32*4096 elements = 256 KB in d_ws.
__global__ void scale_pack_kernel(const float* __restrict__ x,
                                  const float* __restrict__ scales,
                                  _Float16* __restrict__ xs) {
  int idx = blockIdx.x * blockDim.x + threadIdx.x;     // 0 .. TOKENS*IN_F-1
  int i   = idx & (IN_F - 1);
  xs[idx] = (_Float16)(x[idx] * scales[i]);
}

// Kernel 2: out[t, n] = sum_k xs[t,k] * w[n,k]   via v_wmma_f32_16x16x32_f16.
// 4 waves per block; each wave owns one 16-wide N tile and both 16-row M tiles.
__global__ __launch_bounds__(128)
void q8_gemm_wmma_kernel(const _Float16* __restrict__ xs,
                         const int* __restrict__ w,
                         float* __restrict__ out) {
  const int wave  = threadIdx.x >> 5;
  const int lane  = threadIdx.x & 31;
  const int nTile = blockIdx.x * 4 + wave;
  const int n0    = nTile * 16;
  const int nl    = lane & 15;    // column within tile (B/C/D row-stripe)
  const int half  = lane >> 4;    // lane half selects K sub-range

  // A fragment base pointers (ISA 16-bit A 16x32 layout):
  //   VGPR0-3 hold K = half*8 + 0..7, VGPR4-7 hold K = 16 + half*8 + 0..7
  const _Float16* a0p = xs + (size_t)nl * IN_F + half * 8;        // rows 0..15
  const _Float16* a1p = a0p + (size_t)16 * IN_F;                  // rows 16..31
  // B fragment base (ISA 16-bit B 32x16 layout): lane nl = column,
  //   K = k0 + half*8 + 0..7 and k0 + 16 + half*8 + 0..7 from weight row n0+nl.
  const int* wp = w + (size_t)(n0 + nl) * IN_F + half * 8;

  v8f c0 = {};  // M = 0..15
  v8f c1 = {};  // M = 16..31

  for (int k0 = 0; k0 < IN_F; k0 += 32) {
    // ---- A fragments: 2x b128 loads each, already f16 ----
    v16h a0, a1;
    ((v8h*)&a0)[0] = *(const v8h*)(a0p + k0);
    ((v8h*)&a0)[1] = *(const v8h*)(a0p + k0 + 16);
    ((v8h*)&a1)[0] = *(const v8h*)(a1p + k0);
    ((v8h*)&a1)[1] = *(const v8h*)(a1p + k0 + 16);

    // ---- B fragment: 16 int32 as two runs of 8 (64B per lane), 4x b128 loads ----
    v4i w0 = *(const v4i*)(wp + k0);
    v4i w1 = *(const v4i*)(wp + k0 + 4);
    v4i w2 = *(const v4i*)(wp + k0 + 16);
    v4i w3 = *(const v4i*)(wp + k0 + 20);
    // Stream-ahead prefetch of the weight row (~4 iterations / 512 B ahead),
    // WGP scope (locality 3) so it lands in near caches, not just GL2.
    __builtin_prefetch(wp + k0 + 128, 0, 3);

    // int8-valued weights: |v| <= 127, so low 16 bits are a valid i16 and
    // (f16)(i16) is exact -> single v_cvt_f16_i16 per element.
    v16h b;
#pragma unroll
    for (int j = 0; j < 4; ++j) {
      b[j]      = (_Float16)(short)w0[j];
      b[4 + j]  = (_Float16)(short)w1[j];
      b[8 + j]  = (_Float16)(short)w2[j];
      b[12 + j] = (_Float16)(short)w3[j];
    }

    // D = A x B + C   (two M tiles share the streamed B fragment)
    c0 = __builtin_amdgcn_wmma_f32_16x16x32_f16(
        false, a0, false, b, (short)0, c0, false, false);
    c1 = __builtin_amdgcn_wmma_f32_16x16x32_f16(
        false, a1, false, b, (short)0, c1, false, false);
  }

  // C/D layout: VGPR r holds row (half*8 + r), column = n0 + nl.
#pragma unroll
  for (int r = 0; r < 8; ++r) {
    out[(size_t)(half * 8 + r) * OUT_F + n0 + nl]      = c0[r];
    out[(size_t)(16 + half * 8 + r) * OUT_F + n0 + nl] = c1[r];
  }
}

extern "C" void kernel_launch(void* const* d_in, const int* in_sizes, int n_in,
                              void* d_out, int out_size, void* d_ws, size_t ws_size,
                              hipStream_t stream) {
  const float* x      = (const float*)d_in[0];   // [32, 4096] f32
  const int*   weight = (const int*)d_in[1];     // [14336, 4096] int32 (int8-valued)
  const float* scales = (const float*)d_in[2];   // [4096] f32
  float*       out    = (float*)d_out;           // [32, 14336] f32
  _Float16*    xs     = (_Float16*)d_ws;         // 256 KB scratch

  scale_pack_kernel<<<(TOKENS * IN_F) / 256, 256, 0, stream>>>(x, scales, xs);

  // 14336 / 16 = 896 N tiles; 4 waves (tiles) per block -> 224 blocks.
  q8_gemm_wmma_kernel<<<OUT_F / (16 * 4), 128, 0, stream>>>(xs, weight, out);
}
